// LowRankKeyValue_61512521614068
// MI455X (gfx1250) — matrix operand order repaired and checked by reading, not compiled, not run on hardware
//
#include <hip/hip_runtime.h>

// ---- CDNA5 vector types (trivial ext_vectors; safe in unions, match builtin sigs) ----
typedef __attribute__((ext_vector_type(16))) __bf16        v16bf;
typedef __attribute__((ext_vector_type(8)))  float         v8f;
typedef __attribute__((ext_vector_type(4)))  float         f32x4;
typedef __attribute__((ext_vector_type(4)))  unsigned int  u32x4;

#define HIDDEN     2048
#define NTOT       576          // 32 (A_k) + 32 (A_v) + 256 (B_k) + 256 (B_v)
#define NT_TILES   36           // 576 / 16
#define KT_TILES   64           // 2048 / 32
#define LDSTRIDE   584          // 576 + 8 floats pad -> rotates banks per row
#define BLOCK_ROWS 32
#define THREADS    128          // 4 waves; each wave: 2 row-tiles x 9 col-tiles
#define M_ROWS     16384        // B*S
#define OUT_HALF   ((size_t)M_ROWS * 16 * 128)   // elements in k (== in v)
#define WPACK_ELEMS ((size_t)HIDDEN * NTOT)      // 1,179,648
#define APACK_ELEMS ((size_t)M_ROWS * HIDDEN)    // 33,554,432

// fp32 -> bf16 round-to-nearest-even
__device__ __forceinline__ unsigned short bf16_rne(float f) {
  unsigned int u = __builtin_bit_cast(unsigned int, f);
  return (unsigned short)((u + 0x7FFFu + ((u >> 16) & 1u)) >> 16);
}
__device__ __forceinline__ unsigned int pk_bf16(float lo, float hi) {
  unsigned int ulo = __builtin_bit_cast(unsigned int, lo);
  unsigned int uhi = __builtin_bit_cast(unsigned int, hi);
  ulo = (ulo + 0x7FFFu + ((ulo >> 16) & 1u)) >> 16;
  uhi = (uhi + 0x7FFFu + ((uhi >> 16) & 1u)) & 0xFFFF0000u;
  return ulo | uhi;
}

// Inline A-fragment build (fallback path when workspace is too small for apack).
// ISA layout (16-bit A 16x32): lane l: M = l%16; half = l/16;
//   e=0..7  -> K = 8*half + e ; e=8..15 -> K = 16 + 8*half + (e-8)
__device__ __forceinline__ v16bf make_afrag(const float* __restrict__ xrow,
                                            int k0, int lhalf) {
  union { v16bf v; unsigned int u[8]; } af;
  const f32x4 a = *(const f32x4*)(xrow + k0 + 8 * lhalf);
  const f32x4 b = *(const f32x4*)(xrow + k0 + 8 * lhalf + 4);
  const f32x4 c = *(const f32x4*)(xrow + k0 + 16 + 8 * lhalf);
  const f32x4 d = *(const f32x4*)(xrow + k0 + 16 + 8 * lhalf + 4);
  af.u[0] = pk_bf16(a.x, a.y); af.u[1] = pk_bf16(a.z, a.w);
  af.u[2] = pk_bf16(b.x, b.y); af.u[3] = pk_bf16(b.z, b.w);
  af.u[4] = pk_bf16(c.x, c.y); af.u[5] = pk_bf16(c.z, c.w);
  af.u[6] = pk_bf16(d.x, d.y); af.u[7] = pk_bf16(d.z, d.w);
  return af.v;
}

// Pack the 4 fp32 weight matrices into bf16, pre-swizzled into B-fragment order:
// flat idx = ((kt*36 + nt)*32 + lane)*16 + e,  K = kt*32 + 8*(lane/16) + (e<8 ? e : e+8),
// N = nt*16 + lane%16  (ISA layout of 16-bit B 32x16).
__global__ __launch_bounds__(256) void lrkv_pack_w(const float* __restrict__ Wka,
                                                   const float* __restrict__ Wva,
                                                   const float* __restrict__ Wkb,
                                                   const float* __restrict__ Wvb,
                                                   unsigned short* __restrict__ wpack) {
  const int idx = blockIdx.x * 256 + threadIdx.x;   // exactly 2048*576 threads
  const int e  = idx & 15;
  const int l  = (idx >> 4) & 31;
  const int t  = idx >> 9;
  const int nt = t % NT_TILES;
  const int kt = t / NT_TILES;
  const int K  = kt * 32 + ((l >> 4) << 3) + (e < 8 ? e : e + 8);
  const int c  = nt * 16 + (l & 15);
  float val;
  if      (c < 32)  val = Wka[K * 32  + c];
  else if (c < 64)  val = Wva[K * 32  + (c - 32)];
  else if (c < 320) val = Wkb[K * 256 + (c - 64)];
  else              val = Wvb[K * 256 + (c - 320)];
  wpack[idx] = bf16_rne(val);
}

// Pack x into bf16 A-fragment order, one-time memory-bound pass:
// flat idx = ((tr*64 + kt)*32 + lane)*16 + e, row = tr*16 + lane%16,
// K = kt*32 + 8*(lane/16) + (e<8 ? e : e+8).
__global__ __launch_bounds__(256) void lrkv_pack_x(const float* __restrict__ x,
                                                   unsigned short* __restrict__ apack) {
  const size_t idx = (size_t)blockIdx.x * 256 + threadIdx.x;  // 33,554,432 threads
  const int    e  = (int)(idx & 15);
  const int    l  = (int)((idx >> 4) & 31);
  const size_t t  = idx >> 9;
  const int    kt = (int)(t & (KT_TILES - 1));
  const size_t tr = t >> 6;                       // row-tile 0..1023
  const int    K  = kt * 32 + ((l >> 4) << 3) + (e < 8 ? e : e + 8);
  const float  val = __builtin_nontemporal_load(x + (tr * 16 + (l & 15)) * HIDDEN + K);
  apack[idx] = bf16_rne(val);
}

template <bool PRE>
__global__ __launch_bounds__(THREADS) void lrkv_gemm(
    const float* __restrict__ x,
    const unsigned short* __restrict__ apack,
    const unsigned short* __restrict__ wpack,
    float* __restrict__ out) {
  __shared__ __align__(16) float lds[BLOCK_ROWS * LDSTRIDE];   // 74.75 KB

  const int tid   = threadIdx.x;
  const int lane  = tid & 31;
  const int wave  = tid >> 5;       // 0..3 = column group (9 tiles of 16 each)
  const int l16   = lane & 15;
  const int lhalf = lane >> 4;
  const size_t row0 = (size_t)blockIdx.x * BLOCK_ROWS;

  const unsigned short* wbase = wpack + (size_t)wave * 9 * 512 + lane * 16;
  const float* xr0 = nullptr;
  const float* xr1 = nullptr;
  const unsigned short* ap0 = nullptr;
  const unsigned short* ap1 = nullptr;
  if constexpr (PRE) {
    // row-tiles 2*blk and 2*blk+1, fragment-contiguous: 512 elems per (tile,kt)
    ap0 = apack + ((size_t)(blockIdx.x * 2 + 0) * KT_TILES) * 512 + lane * 16;
    ap1 = apack + ((size_t)(blockIdx.x * 2 + 1) * KT_TILES) * 512 + lane * 16;
  } else {
    xr0 = x + (row0 + l16) * HIDDEN;
    xr1 = x + (row0 + 16 + l16) * HIDDEN;
  }

  v8f acc[2][9] = {};   // 2 row-tiles x 9 col-tiles, f32 accum (144 VGPRs)

  for (int kt = 0; kt < KT_TILES; ++kt) {
    v16bf a0, a1;
    if constexpr (PRE) {
      union { v16bf v; u32x4 q[2]; } t0, t1;
      const u32x4* p0 = (const u32x4*)(ap0 + (size_t)kt * 512);
      const u32x4* p1 = (const u32x4*)(ap1 + (size_t)kt * 512);
      t0.q[0] = p0[0]; t0.q[1] = p0[1];
      t1.q[0] = p1[0]; t1.q[1] = p1[1];
      a0 = t0.v; a1 = t1.v;
    } else {
      const int k0 = kt * 32;
      a0 = make_afrag(xr0, k0, lhalf);
      a1 = make_afrag(xr1, k0, lhalf);
    }
    const unsigned short* wk = wbase + (size_t)kt * NT_TILES * 512;
#pragma unroll
    for (int j = 0; j < 9; ++j) {
      union { v16bf v; u32x4 q[2]; } bfr;
      const u32x4* bp = (const u32x4*)(wk + j * 512);
      bfr.q[0] = bp[0];
      bfr.q[1] = bp[1];
      acc[0][j] = __builtin_amdgcn_wmma_f32_16x16x32_bf16(
          false, a0, false, bfr.v, (short)0, acc[0][j], false, false);
      acc[1][j] = __builtin_amdgcn_wmma_f32_16x16x32_bf16(
          false, a1, false, bfr.v, (short)0, acc[1][j], false, false);
    }
  }

  // C layout (f32 16x16): lane l, vgpr v -> M = v + 8*(l/16), N = l%16
#pragma unroll
  for (int rt = 0; rt < 2; ++rt) {
#pragma unroll
    for (int j = 0; j < 9; ++j) {
      const int col = (wave * 9 + j) * 16 + l16;
#pragma unroll
      for (int vv = 0; vv < 8; ++vv) {
        const int m = rt * 16 + vv + 8 * lhalf;
        lds[m * LDSTRIDE + col] = acc[rt][j][vv];
      }
    }
  }
  __syncthreads();

  // Rank-2 combine: k[h,d] = 0.5*(A_k[h,0]*B_k[0,d] + A_k[h,1]*B_k[1,d]); same for v.
  // Columns in LDS row: [0,32)=A_k, [32,64)=A_v, [64,320)=B_k, [320,576)=B_v.
  float* kout = out;
  float* vout = out + OUT_HALF;
#pragma unroll 1
  for (int it = 0; it < 128; ++it) {          // 32 rows * 16 h * 32 float4 / 128 thr
    const int idx = it * THREADS + tid;
    const int dv  = idx & 31;                 // float4 index along d
    const int h   = (idx >> 5) & 15;
    const int r   = idx >> 9;                 // 0..31
    const float* lr = lds + r * LDSTRIDE;
    const float ak0 = lr[2 * h],      ak1 = lr[2 * h + 1];
    const float av0 = lr[32 + 2 * h], av1 = lr[32 + 2 * h + 1];
    const f32x4 bk0 = *(const f32x4*)(lr + 64  + dv * 4);
    const f32x4 bk1 = *(const f32x4*)(lr + 192 + dv * 4);
    const f32x4 bv0 = *(const f32x4*)(lr + 320 + dv * 4);
    const f32x4 bv1 = *(const f32x4*)(lr + 448 + dv * 4);
    const f32x4 ko = 0.5f * (ak0 * bk0 + ak1 * bk1);
    const f32x4 vo = 0.5f * (av0 * bv0 + av1 * bv1);
    const size_t g = ((row0 + r) * 16 + h) * 128 + dv * 4;
    // Outputs are streamed and never re-read: keep them out of the caches.
    __builtin_nontemporal_store(ko, (f32x4*)(kout + g));
    __builtin_nontemporal_store(vo, (f32x4*)(vout + g));
  }
}

extern "C" void kernel_launch(void* const* d_in, const int* in_sizes, int n_in,
                              void* d_out, int out_size, void* d_ws, size_t ws_size,
                              hipStream_t stream) {
  const float* x   = (const float*)d_in[0];
  const float* Wka = (const float*)d_in[1];
  const float* Wva = (const float*)d_in[2];
  const float* Wkb = (const float*)d_in[3];
  const float* Wvb = (const float*)d_in[4];
  unsigned short* wpack = (unsigned short*)d_ws;   // 2.36 MB
  float* out = (float*)d_out;

  lrkv_pack_w<<<(int)(WPACK_ELEMS / 256), 256, 0, stream>>>(Wka, Wva, Wkb, Wvb, wpack);

  const size_t need = (WPACK_ELEMS + APACK_ELEMS) * sizeof(unsigned short); // ~69.5 MB
  if (ws_size >= need) {
    // Fast path: one-time bf16 pre-pack of x into A-fragment layout, then a lean
    // WMMA loop (4 A-loads + 18 B-loads + 18 WMMAs per k-step, no conversion VALU).
    unsigned short* apack = wpack + WPACK_ELEMS;
    lrkv_pack_x<<<(int)(APACK_ELEMS / 256), 256, 0, stream>>>(x, apack);
    lrkv_gemm<true><<<M_ROWS / BLOCK_ROWS, THREADS, 0, stream>>>(x, apack, wpack, out);
  } else {
    // Fallback: convert x to bf16 fragments in-kernel.
    lrkv_gemm<false><<<M_ROWS / BLOCK_ROWS, THREADS, 0, stream>>>(x, nullptr, wpack, out);
  }
}
